// MultiHeadAttention_89043261981199
// MI455X (gfx1250) — matrix operand order repaired and checked by reading, not compiled, not run on hardware
//
#include <hip/hip_runtime.h>
#include <hip/hip_bf16.h>

// MI455X (gfx1250) mixed-precision MHA:
//   - all matmuls via v_wmma_f32_16x16x32_bf16 (fp32 accumulate)
//   - GEMM bf16 tiles staged with GLOBAL_LOAD_ASYNC_TO_LDS_B128 (+ s_wait_asynccnt)
//   - softmax reductions via DPP16 butterflies (no LDS shuffles)
//   - 1/sqrt(Dh) folded into the Q projection epilogue (power-of-two, exact)
//   - flash-style streaming softmax attention, 32 queries/wave, P staged via LDS
// Workspace layout (assumes ws_size >= ~28 MB for L=2048, D=1024):
//   [4 x D*D bf16 transposed weights][Qb][Kb][Vb][Vt][AttnOut]  (each L*D bf16)

typedef __attribute__((ext_vector_type(16))) __bf16 v16bf;
typedef __attribute__((ext_vector_type(8)))  __bf16 v8bf;
typedef __attribute__((ext_vector_type(8)))  float  v8f;

__device__ __forceinline__ v8f wmma_bf16(v16bf a, v16bf b, v8f c) {
  return __builtin_amdgcn_wmma_f32_16x16x32_bf16(false, a, false, b, (short)0, c,
                                                 false, false);
}

// ---- DPP16 16-lane butterfly reductions (all-VALU, no LDS) -----------------
template <int CTRL>
__device__ __forceinline__ float dpp_mov(float x) {
  return __builtin_bit_cast(
      float, __builtin_amdgcn_update_dpp(0, __builtin_bit_cast(int, x), CTRL,
                                         0xF, 0xF, true));
}
__device__ __forceinline__ float red16_max(float x) {
  x = fmaxf(x, dpp_mov<0xB1>(x));   // quad_perm(1,0,3,2)  : xor 1
  x = fmaxf(x, dpp_mov<0x4E>(x));   // quad_perm(2,3,0,1)  : xor 2
  x = fmaxf(x, dpp_mov<0x141>(x));  // row_half_mirror     : xor 4 (post-fold)
  x = fmaxf(x, dpp_mov<0x140>(x));  // row_mirror          : xor 8 (post-fold)
  return x;
}
__device__ __forceinline__ float red16_sum(float x) {
  x += dpp_mov<0xB1>(x);
  x += dpp_mov<0x4E>(x);
  x += dpp_mov<0x141>(x);
  x += dpp_mov<0x140>(x);
  return x;
}

// A-fragment (16x32 bf16): lane: M = lane&15,
// K = (lane>>4)*16 + e (16 contiguous). `ld` in halves (mult of 8).
__device__ __forceinline__ v16bf load_a_frag(const __bf16* base, int ld, int lane) {
  const __bf16* p = base + (size_t)(lane & 15) * ld + ((lane >> 4) << 4);
  v8bf lo = *(const v8bf*)p;
  v8bf hi = *(const v8bf*)(p + 8);
  return __builtin_shufflevector(lo, hi, 0, 1, 2, 3, 4, 5, 6, 7,
                                 8, 9, 10, 11, 12, 13, 14, 15);
}

// B-fragment (32x16 bf16): lane: N = lane&15, K = (lane>>4)*16 + e (16
// contiguous). Source stored as [n][k] rows, `ld` in halves (multiple of 16).
__device__ __forceinline__ v16bf load_b_frag(const __bf16* base, int ld, int lane) {
  const __bf16* p = base + (size_t)(lane & 15) * ld + ((lane >> 4) << 4);
  return *(const v16bf*)p;
}

// Async global->LDS 16B copy (per-lane addresses), tracked on ASYNCcnt.
__device__ __forceinline__ void async_copy_b128(unsigned lds_off, const void* gsrc) {
  unsigned long long ga = (unsigned long long)gsrc;
  asm volatile("global_load_async_to_lds_b128 %0, %1, off"
               :: "v"(lds_off), "v"(ga)
               : "memory");
}
__device__ __forceinline__ void wait_async0() {
  asm volatile("s_wait_asynccnt 0x0" ::: "memory");
}

// ---------------------------------------------------------------------------
// Weight prep: fp32 W[K][N]  ->  bf16 Wt[N][K]
// ---------------------------------------------------------------------------
__global__ __launch_bounds__(256) void transpose_w_kernel(const float* __restrict__ W,
                                                          __bf16* __restrict__ Wt,
                                                          int K, int N) {
  __shared__ float tile[32][33];
  int tx = threadIdx.x & 31, ty = threadIdx.x >> 5;  // 32x8
  int bk = blockIdx.x * 32, bn = blockIdx.y * 32;
#pragma unroll
  for (int i = 0; i < 32; i += 8)
    tile[ty + i][tx] = W[(size_t)(bk + ty + i) * N + bn + tx];
  __syncthreads();
#pragma unroll
  for (int i = 0; i < 32; i += 8)
    Wt[(size_t)(bn + ty + i) * K + bk + tx] = (__bf16)tile[tx][ty + i];
}

// bf16 [R][C] -> bf16 [C][R]   (V -> V^T so attention's V B-fragments are contiguous)
__global__ __launch_bounds__(256) void transpose_bf_kernel(const __bf16* __restrict__ S,
                                                           __bf16* __restrict__ Dst,
                                                           int R, int C) {
  __shared__ __bf16 tile[32][34];
  int tx = threadIdx.x & 31, ty = threadIdx.x >> 5;
  int br = blockIdx.x * 32, bc = blockIdx.y * 32;
#pragma unroll
  for (int i = 0; i < 32; i += 8)
    tile[ty + i][tx] = S[(size_t)(br + ty + i) * C + bc + tx];
  __syncthreads();
#pragma unroll
  for (int i = 0; i < 32; i += 8)
    Dst[(size_t)(bc + ty + i) * R + br + tx] = tile[tx][ty + i];
}

// ---------------------------------------------------------------------------
// GEMM + bias + output scale:  C[M][N] = ((A[M][K] * W[K][N]) + bias[N]) * oscale
// A fp32 (VALU-converted) or bf16 (async-to-LDS); Wt bf16 [N][K] (async-to-LDS).
// Block tile 128x64, 8 waves, wave tile 32x32 (2x2 WMMA), k-step 32.
// ---------------------------------------------------------------------------
template <bool A_BF16, bool OUT_BF16>
__global__ __launch_bounds__(256) void gemm_bias_kernel(const void* __restrict__ Aptr,
                                                        const __bf16* __restrict__ Wt,
                                                        const float* __restrict__ bias,
                                                        void* __restrict__ Cptr,
                                                        int M, int N, int K,
                                                        float oscale) {
  constexpr int LDA = 40;                       // halves; 80B rows (16B aligned)
  constexpr int LDB = 48;                       // halves; 96B rows (32B aligned)
  constexpr unsigned A_BYTES = 128 * LDA * 2;   // 10240
  constexpr unsigned B_OFF   = A_BYTES;         // Bs starts here (32B aligned)
  __shared__ __align__(128) char smem[A_BYTES + 64 * LDB * 2];
  __bf16* As = (__bf16*)smem;
  __bf16* Bs = (__bf16*)(smem + B_OFF);

  const int tid  = threadIdx.x;
  const int lane = tid & 31;
  const int wave = tid >> 5;
  const int bm = blockIdx.x * 128;
  const int bn = blockIdx.y * 64;
  const int wm = (wave >> 1) * 32;
  const int wn = (wave & 1) * 32;

  v8f acc[2][2] = {};

  for (int k0 = 0; k0 < K; k0 += 32) {
    // ---- stage B tile (64 x 32) async: 16B per lane ----
    {
      int rb  = tid >> 2;
      int kob = (tid & 3) * 8;
      unsigned dst = B_OFF + (unsigned)(rb * LDB + kob) * 2;
      async_copy_b128(dst, Wt + (size_t)(bn + rb) * K + k0 + kob);
    }
    // ---- stage A tile (128 x 32) ----
    {
      int r  = tid >> 1;
      int ko = (tid & 1) * 16;
      if (A_BF16) {
        unsigned dst = (unsigned)(r * LDA + ko) * 2;
        const __bf16* src = (const __bf16*)Aptr + (size_t)(bm + r) * K + k0 + ko;
        async_copy_b128(dst, src);
        async_copy_b128(dst + 16u, src + 8);
      } else {
        const float* arow = (const float*)Aptr + (size_t)(bm + r) * K + k0 + ko;
        if (k0 + 32 < K) __builtin_prefetch(arow + 32, 0, 1);
        const float4* ap = (const float4*)arow;
        float4 f0 = ap[0], f1 = ap[1], f2 = ap[2], f3 = ap[3];
        v8bf b0, b1;
        b0[0] = (__bf16)f0.x; b0[1] = (__bf16)f0.y; b0[2] = (__bf16)f0.z; b0[3] = (__bf16)f0.w;
        b0[4] = (__bf16)f1.x; b0[5] = (__bf16)f1.y; b0[6] = (__bf16)f1.z; b0[7] = (__bf16)f1.w;
        b1[0] = (__bf16)f2.x; b1[1] = (__bf16)f2.y; b1[2] = (__bf16)f2.z; b1[3] = (__bf16)f2.w;
        b1[4] = (__bf16)f3.x; b1[5] = (__bf16)f3.y; b1[6] = (__bf16)f3.z; b1[7] = (__bf16)f3.w;
        *(v8bf*)&As[r * LDA + ko]     = b0;
        *(v8bf*)&As[r * LDA + ko + 8] = b1;
      }
    }
    wait_async0();
    __syncthreads();

    v16bf a0 = load_a_frag(As + wm * LDA,        LDA, lane);
    v16bf a1 = load_a_frag(As + (wm + 16) * LDA, LDA, lane);
    v16bf b0 = load_b_frag(Bs + wn * LDB,        LDB, lane);
    v16bf b1 = load_b_frag(Bs + (wn + 16) * LDB, LDB, lane);

    acc[0][0] = wmma_bf16(a0, b0, acc[0][0]);
    acc[0][1] = wmma_bf16(a0, b1, acc[0][1]);
    acc[1][0] = wmma_bf16(a1, b0, acc[1][0]);
    acc[1][1] = wmma_bf16(a1, b1, acc[1][1]);
    __syncthreads();
  }

  // ---- epilogue: C/D layout M = v + 8*(lane>>4), N = lane&15 ----
  const int hi   = lane >> 4;
  const int col0 = lane & 15;
#pragma unroll
  for (int mi = 0; mi < 2; ++mi) {
#pragma unroll
    for (int ni = 0; ni < 2; ++ni) {
      int colg = bn + wn + ni * 16 + col0;
      float bv = bias[colg];
#pragma unroll
      for (int v = 0; v < 8; ++v) {
        int rowg  = bm + wm + mi * 16 + v + 8 * hi;
        float val = (acc[mi][ni][v] + bv) * oscale;
        if (OUT_BF16)
          ((__bf16*)Cptr)[(size_t)rowg * N + colg] = (__bf16)val;
        else
          ((float*)Cptr)[(size_t)rowg * N + colg] = val;
      }
    }
  }
}

// ---------------------------------------------------------------------------
// Flash-style attention. One wave = 32 queries x one head (Dh = 64).
// Q is pre-scaled by 1/sqrt(Dh) at projection time, so the inner loop is
// wmma + softmax only. Streams keys/values in tiles of 32; online softmax
// (DPP reductions); 16 WMMAs per key tile; K/V loads amortized over both
// query tiles. Qb/Kb: bf16 [L][1024]; Vt: bf16 [1024][L]; Ob: bf16 [L][1024].
// ---------------------------------------------------------------------------
__global__ __launch_bounds__(128) void attention_kernel(const __bf16* __restrict__ Qb,
                                                        const __bf16* __restrict__ Kb,
                                                        const __bf16* __restrict__ Vt,
                                                        __bf16* __restrict__ Ob,
                                                        int L) {
  constexpr int D   = 1024;
  constexpr int DH  = 64;
  constexpr int LDP = 48;            // halves
  __shared__ __bf16 Ps[4][32][LDP];  // per-wave P staging (32 x 32)

  const int lane   = threadIdx.x & 31;
  const int wave   = threadIdx.x >> 5;
  const int qtiles = L >> 5;
  const int task   = blockIdx.x * 4 + wave;
  const int h      = task / qtiles;
  const int q0     = (task % qtiles) * 32;

  v16bf qa[2][2];
#pragma unroll
  for (int qt = 0; qt < 2; ++qt)
#pragma unroll
    for (int c = 0; c < 2; ++c)
      qa[qt][c] = load_a_frag(Qb + (size_t)(q0 + qt * 16) * D + h * DH + c * 32,
                              D, lane);

  float m_i[2][8], l_i[2][8];
  v8f o[2][4] = {};
#pragma unroll
  for (int qt = 0; qt < 2; ++qt)
#pragma unroll
    for (int v = 0; v < 8; ++v) { m_i[qt][v] = -1e30f; l_i[qt][v] = 0.f; }

  for (int kt = 0; kt < L; kt += 32) {
    // ---- S = Q * K^T for 2 query tiles x 2 key tiles ----
    const __bf16* kbase = Kb + (size_t)kt * D + h * DH;
    v16bf kb[2][2];
    kb[0][0] = load_b_frag(kbase,               D, lane);
    kb[0][1] = load_b_frag(kbase + 32,          D, lane);
    kb[1][0] = load_b_frag(kbase + 16 * D,      D, lane);
    kb[1][1] = load_b_frag(kbase + 16 * D + 32, D, lane);

    v8f s[2][2];
#pragma unroll
    for (int qt = 0; qt < 2; ++qt)
#pragma unroll
      for (int j = 0; j < 2; ++j) {
        v8f z = {};
        z = wmma_bf16(qa[qt][0], kb[j][0], z);
        z = wmma_bf16(qa[qt][1], kb[j][1], z);
        s[qt][j] = z;
      }

    // ---- online softmax (16-lane DPP reductions per row) ----
    float alpha[2][8];
#pragma unroll
    for (int qt = 0; qt < 2; ++qt)
#pragma unroll
      for (int v = 0; v < 8; ++v) {
        float a = s[qt][0][v], b = s[qt][1][v];
        float mx = red16_max(fmaxf(a, b));
        float mn = fmaxf(m_i[qt][v], mx);
        alpha[qt][v] = __expf(m_i[qt][v] - mn);
        m_i[qt][v]   = mn;
        float p0 = __expf(a - mn), p1 = __expf(b - mn);
        s[qt][0][v] = p0;
        s[qt][1][v] = p1;
        l_i[qt][v] = l_i[qt][v] * alpha[qt][v] + red16_sum(p0 + p1);
      }
#pragma unroll
    for (int qt = 0; qt < 2; ++qt)
#pragma unroll
      for (int n = 0; n < 4; ++n)
#pragma unroll
        for (int v = 0; v < 8; ++v) o[qt][n][v] *= alpha[qt][v];

    // ---- stage P (32x32) to LDS, reload as A-fragments ----
    {
      int col = lane & 15, rb = (lane >> 4) * 8;
#pragma unroll
      for (int qt = 0; qt < 2; ++qt)
#pragma unroll
        for (int v = 0; v < 8; ++v) {
          Ps[wave][qt * 16 + rb + v][col]      = (__bf16)s[qt][0][v];
          Ps[wave][qt * 16 + rb + v][col + 16] = (__bf16)s[qt][1][v];
        }
    }
    asm volatile("s_wait_dscnt 0x0" ::: "memory");
    v16bf pa0 = load_a_frag(&Ps[wave][0][0],  LDP, lane);
    v16bf pa1 = load_a_frag(&Ps[wave][16][0], LDP, lane);

    // ---- O += P * V  (V^T rows are dh, contiguous over keys) ----
    const __bf16* vt = Vt + (size_t)h * DH * L + kt;
#pragma unroll
    for (int n = 0; n < 4; ++n) {
      v16bf vb = load_b_frag(vt + (size_t)(n * 16) * L, L, lane);
      o[0][n] = wmma_bf16(pa0, vb, o[0][n]);
      o[1][n] = wmma_bf16(pa1, vb, o[1][n]);
    }
  }

  // ---- normalize and store (concat-head layout) ----
  const int hi = lane >> 4, col = lane & 15;
#pragma unroll
  for (int qt = 0; qt < 2; ++qt) {
    __bf16* obase = Ob + (size_t)(q0 + qt * 16) * D + h * DH;
#pragma unroll
    for (int n = 0; n < 4; ++n)
#pragma unroll
      for (int v = 0; v < 8; ++v) {
        float val = o[qt][n][v] / l_i[qt][v];
        obase[(size_t)(v + 8 * hi) * D + n * 16 + col] = (__bf16)val;
      }
  }
}

// ---------------------------------------------------------------------------
extern "C" void kernel_launch(void* const* d_in, const int* in_sizes, int n_in,
                              void* d_out, int out_size, void* d_ws, size_t ws_size,
                              hipStream_t stream) {
  (void)n_in; (void)out_size; (void)ws_size;
  const float* q   = (const float*)d_in[0];
  const float* k   = (const float*)d_in[1];
  const float* v   = (const float*)d_in[2];
  const float* w_q = (const float*)d_in[3];
  const float* b_q = (const float*)d_in[4];
  const float* w_k = (const float*)d_in[5];
  const float* b_k = (const float*)d_in[6];
  const float* w_v = (const float*)d_in[7];
  const float* b_v = (const float*)d_in[8];
  const float* w_o = (const float*)d_in[9];
  const float* b_o = (const float*)d_in[10];

  const int D = 1024, NH = 16;
  const int L = in_sizes[0] / D;  // 2048

  __bf16* Wt_q = (__bf16*)d_ws;
  __bf16* Wt_k = Wt_q + (size_t)D * D;
  __bf16* Wt_v = Wt_k + (size_t)D * D;
  __bf16* Wt_o = Wt_v + (size_t)D * D;
  __bf16* Qb   = Wt_o + (size_t)D * D;
  __bf16* Kb   = Qb + (size_t)L * D;
  __bf16* Vb   = Kb + (size_t)L * D;
  __bf16* Vt   = Vb + (size_t)L * D;
  __bf16* Ao   = Vt + (size_t)L * D;

  dim3 tb(256);
  dim3 tgw(D / 32, D / 32);
  transpose_w_kernel<<<tgw, tb, 0, stream>>>(w_q, Wt_q, D, D);
  transpose_w_kernel<<<tgw, tb, 0, stream>>>(w_k, Wt_k, D, D);
  transpose_w_kernel<<<tgw, tb, 0, stream>>>(w_v, Wt_v, D, D);
  transpose_w_kernel<<<tgw, tb, 0, stream>>>(w_o, Wt_o, D, D);

  dim3 gg(L / 128, D / 64);
  // Q projection carries the 1/sqrt(Dh) = 0.125 attention scale (exact in bf16).
  gemm_bias_kernel<false, true><<<gg, tb, 0, stream>>>(q, Wt_q, b_q, Qb, L, D, D,
                                                       0.125f);
  gemm_bias_kernel<false, true><<<gg, tb, 0, stream>>>(k, Wt_k, b_k, Kb, L, D, D,
                                                       1.0f);
  gemm_bias_kernel<false, true><<<gg, tb, 0, stream>>>(v, Wt_v, b_v, Vb, L, D, D,
                                                       1.0f);

  transpose_bf_kernel<<<dim3(L / 32, D / 32), tb, 0, stream>>>(Vb, Vt, L, D);

  int tasks = NH * (L / 32);  // 1024 wave-tasks
  attention_kernel<<<tasks / 4, 128, 0, stream>>>(Qb, Kb, Vt, Ao, L);

  gemm_bias_kernel<true, false><<<gg, tb, 0, stream>>>(Ao, Wt_o, b_o, (float*)d_out,
                                                       L, D, D, 1.0f);
}
